// GCNMerge_76244259438916
// MI455X (gfx1250) — matrix operand order repaired and checked by reading, not compiled, not run on hardware
//
#include <hip/hip_runtime.h>

#define NN   8192
#define NE   524288
#define IND  256
#define H1   128
#define H2   64
#define ETOT (NE + NN)

typedef __attribute__((ext_vector_type(16))) _Float16 v16h;
typedef __attribute__((ext_vector_type(8)))  _Float16 v8h;
typedef __attribute__((ext_vector_type(8)))  float    v8f;

// ---------------- init accumulators ----------------
__global__ void k_init(float* deg, float* hc, float* mu, float* lv) {
    int i = blockIdx.x * blockDim.x + threadIdx.x;
    int stride = gridDim.x * blockDim.x;
    for (int t = i; t < NN; t += stride) deg[t] = 1.0f;            // self-loop contributes 1
    for (int t = i; t < NN * 2 * H1; t += stride) hc[t] = 0.0f;
    for (int t = i; t < NN * H2; t += stride) { mu[t] = 0.0f; lv[t] = 0.0f; }
}

// ---------------- degree ----------------
__global__ void k_deg(const int* __restrict__ ei, float* deg) {
    int e = blockIdx.x * blockDim.x + threadIdx.x;
    if (e < NE) atomicAdd(&deg[ei[e]], 1.0f);
}

__global__ void k_dinv(float* deg) {
    int i = blockIdx.x * blockDim.x + threadIdx.x;
    if (i < NN) { float d = deg[i]; deg[i] = (d > 0.0f) ? rsqrtf(d) : 0.0f; }
}

// ---------------- GEMM1: h = x @ [W1|W2]  (8192x256 @ 256x256) ----------------
__global__ void k_gemm1(const float* __restrict__ x, const float* __restrict__ W1,
                        const float* __restrict__ W2, float* __restrict__ h) {
    __shared__ float xs[IND];
    int row = blockIdx.x;
    int j = threadIdx.x;                    // 0..255
    xs[j] = __builtin_nontemporal_load(x + row * IND + j);   // x read exactly once
    __syncthreads();
    const float* W = (j < H1) ? W1 : W2;
    int jj = j & (H1 - 1);
    float acc = 0.0f;
#pragma unroll 8
    for (int k = 0; k < IND; ++k) acc = fmaf(xs[k], W[k * H1 + jj], acc);
    h[row * (2 * H1) + j] = acc;
}

// ---------------- agg1: hc[row] += norm * h[col]  (256 floats, 1 wave/edge) ----------------
__global__ void k_agg1(const int* __restrict__ ei, const float* __restrict__ dinv,
                       const float* __restrict__ h, float* __restrict__ hc) {
    int wid  = (blockIdx.x * blockDim.x + threadIdx.x) >> 5;
    int lane = threadIdx.x & 31;
    if (wid >= ETOT) return;
    int r, c;
    if (wid < NE) { r = ei[wid]; c = ei[NE + wid]; }
    else          { r = c = wid - NE; }                 // self-loop
    float w = dinv[r] * dinv[c];
    const float4* src = (const float4*)(h + (size_t)c * (2 * H1)) + lane * 2;
    float4 a = src[0], b = src[1];
    float* dst = hc + (size_t)r * (2 * H1) + lane * 8;
    atomicAdd(dst + 0, a.x * w); atomicAdd(dst + 1, a.y * w);
    atomicAdd(dst + 2, a.z * w); atomicAdd(dst + 3, a.w * w);
    atomicAdd(dst + 4, b.x * w); atomicAdd(dst + 5, b.y * w);
    atomicAdd(dst + 6, b.z * w); atomicAdd(dst + 7, b.w * w);
}

// ---------------- GEMM2: g = (hc + [b1|b2]) @ [Wmu|Wlv]  (8192x256 @ 256x128) ----------------
__global__ void k_gemm2(const float* __restrict__ hc, const float* __restrict__ b1,
                        const float* __restrict__ b2, const float* __restrict__ Wmu,
                        const float* __restrict__ Wlv, float* __restrict__ g) {
    __shared__ float xs[2 * H1];
    int row = blockIdx.x;
    int j = threadIdx.x;                    // 0..127
    xs[j]        = hc[row * (2 * H1) + j] + b1[j];
    xs[j + H1]   = hc[row * (2 * H1) + j + H1] + b2[j];
    __syncthreads();
    const float* W = (j < H2) ? Wmu : Wlv;
    int jj = (j < H2) ? j : (j - H2);
    float acc = 0.0f;
#pragma unroll 8
    for (int k = 0; k < 2 * H1; ++k) acc = fmaf(xs[k], W[k * H2 + jj], acc);
    g[row * (2 * H2) + j] = acc;
}

// ---------------- agg2: mu/lv[row] += norm * g[col]  (128 floats, 1 wave/edge) ----------------
__global__ void k_agg2(const int* __restrict__ ei, const float* __restrict__ dinv,
                       const float* __restrict__ g, float* __restrict__ mu,
                       float* __restrict__ lv) {
    int wid  = (blockIdx.x * blockDim.x + threadIdx.x) >> 5;
    int lane = threadIdx.x & 31;
    if (wid >= ETOT) return;
    int r, c;
    if (wid < NE) { r = ei[wid]; c = ei[NE + wid]; }
    else          { r = c = wid - NE; }
    float w = dinv[r] * dinv[c];
    float4 a = *((const float4*)(g + (size_t)c * (2 * H2)) + lane);
    int d = lane * 4;                       // 0..124
    float* dst = (d < H2) ? (mu + (size_t)r * H2 + d)
                          : (lv + (size_t)r * H2 + (d - H2));
    atomicAdd(dst + 0, a.x * w); atomicAdd(dst + 1, a.y * w);
    atomicAdd(dst + 2, a.z * w); atomicAdd(dst + 3, a.w * w);
}

// ---------------- bias + f16 conversion of z ----------------
__global__ void k_finz(float* mu, float* lv, const float* __restrict__ bmu,
                       const float* __restrict__ blv, _Float16* __restrict__ zh) {
    int t = blockIdx.x * blockDim.x + threadIdx.x;
    if (t < NN * H2) {
        int d = t & (H2 - 1);
        float m = mu[t] + bmu[d];
        mu[t] = m;
        zh[t] = (_Float16)m;
        lv[t] = lv[t] + blv[d];
    }
}

// ---------------- adj = sigmoid(z @ z.T) via v_wmma_f32_16x16x32_f16 ----------------
// Each wave: one 16-row tile of A held in registers, sweeps 8 consecutive 16x16
// N-tiles. K=64 -> 2 WMMAs per tile.
// A layout chosen here (contiguous k, not the ISA table order): lanes 0-15 hold M=lane, K=kh..kh+15, kh=0;
//                            lanes 16-31 same rows with kh=16.
// B layout, same contiguous k as A: lanes 0-15 hold N=lane, K=0..15; lanes 16-31 K=16..31.
__global__ void k_adj(const _Float16* __restrict__ zh, float* __restrict__ adj) {
    const int NT = 8;
    int wid  = (blockIdx.x * blockDim.x + threadIdx.x) >> 5;  // 0..32767
    int lane = threadIdx.x & 31;
    int tm = wid >> 6;                       // 0..511  (row tile)
    int tg = wid & 63;                       // 0..63   (group of 8 col tiles)
    int m0 = tm * 16;
    int n0 = tg * (16 * NT);
    int lr = lane & 15;
    bool hi = (lane >= 16);

    // ---- load A (rows m0..m0+15 of z), both K halves ----
    int kh = hi ? 16 : 0;
    const _Float16* ar = zh + (size_t)(m0 + lr) * H2 + kh;
    v8h a0 = *(const v8h*)(ar);        // K kh..kh+7
    v8h a1 = *(const v8h*)(ar + 8);    // K kh+8..kh+15
    v8h a2 = *(const v8h*)(ar + 32);   // K kh+32..kh+39
    v8h a3 = *(const v8h*)(ar + 40);   // K kh+40..kh+47
    v16h alo = __builtin_shufflevector(a0, a1, 0,1,2,3,4,5,6,7,8,9,10,11,12,13,14,15);
    v16h ahi = __builtin_shufflevector(a2, a3, 0,1,2,3,4,5,6,7,8,9,10,11,12,13,14,15);

    int kb = hi ? 16 : 0;
    int rbase = m0 + (hi ? 8 : 0);
#pragma unroll
    for (int nt = 0; nt < NT; ++nt) {
        int col = n0 + nt * 16 + lr;
        const _Float16* br = zh + (size_t)col * H2 + kb;
        v16h blo = *(const v16h*)(br);        // K kb..kb+15
        v16h bhi = *(const v16h*)(br + 32);   // K kb+32..kb+47
        v8f c = {};
        c = __builtin_amdgcn_wmma_f32_16x16x32_f16(false, alo, false, blo,
                                                   (short)0, c, false, false);
        c = __builtin_amdgcn_wmma_f32_16x16x32_f16(false, ahi, false, bhi,
                                                   (short)0, c, false, false);
#pragma unroll
        for (int v = 0; v < 8; ++v) {
            float s = 1.0f / (1.0f + __expf(-c[v]));
            // 256 MB write-once stream: non-temporal so it doesn't evict the
            // L2-resident working set (zh, h, hc).
            __builtin_nontemporal_store(s, adj + (size_t)(rbase + v) * NN + col);
        }
    }
}

// ---------------- launch ----------------
extern "C" void kernel_launch(void* const* d_in, const int* in_sizes, int n_in,
                              void* d_out, int out_size, void* d_ws, size_t ws_size,
                              hipStream_t stream) {
    const float* x   = (const float*)d_in[0];
    const int*   ei  = (const int*)  d_in[1];   // [2, NE] int32 (JAX demotes int64)
    const float* W1  = (const float*)d_in[2];
    const float* b1  = (const float*)d_in[3];
    const float* W2  = (const float*)d_in[4];
    const float* b2  = (const float*)d_in[5];
    const float* Wmu = (const float*)d_in[6];
    const float* bmu = (const float*)d_in[7];
    const float* Wlv = (const float*)d_in[8];
    const float* blv = (const float*)d_in[9];

    float* out = (float*)d_out;
    float* adj = out;                                  // NN*NN
    float* mu  = out + (size_t)NN * NN;                // NN*H2
    float* lv  = mu + (size_t)NN * H2;                 // NN*H2

    float* deg = (float*)d_ws;                         // NN (becomes dinv)
    float* h   = deg + NN;                             // NN * 2*H1
    float* hc  = h   + (size_t)NN * 2 * H1;            // NN * 2*H1
    float* g   = hc  + (size_t)NN * 2 * H1;            // NN * 2*H2
    _Float16* zh = (_Float16*)(g + (size_t)NN * 2 * H2); // NN * H2 f16

    k_init <<<4096, 256, 0, stream>>>(deg, hc, mu, lv);
    k_deg  <<<NE / 256, 256, 0, stream>>>(ei, deg);
    k_dinv <<<NN / 256, 256, 0, stream>>>(deg);
    k_gemm1<<<NN, 256, 0, stream>>>(x, W1, W2, h);
    k_agg1 <<<ETOT / 8, 256, 0, stream>>>(ei, deg, h, hc);
    k_gemm2<<<NN, 128, 0, stream>>>(hc, b1, b2, Wmu, Wlv, g);
    k_agg2 <<<ETOT / 8, 256, 0, stream>>>(ei, deg, g, mu, lv);
    k_finz <<<(NN * H2) / 256, 256, 0, stream>>>(mu, lv, bmu, blv, zh);
    k_adj  <<<4096, 256, 0, stream>>>(zh, adj);
}
